// Nets_71554155151899
// MI455X (gfx1250) — matrix operand order repaired and checked by reading, not compiled, not run on hardware
//
#include <hip/hip_runtime.h>

// ---------------------------------------------------------------------------
// CDNA5 (gfx1250) implementation of the equivariant GNN forward pass.
// Big GEMMs (edge message 608->240, node update 1200->{240,512}) use
// v_wmma_f32_16x16x32_bf16 with f32 accumulation. A-tiles are gathered into
// LDS (f32 -> bf16 on the fly, read back as 2x ds_load_b128 per fragment).
// B weights are pre-repacked into the exact WMMA lane layout so each lane
// loads its fragment with one 32B vector load (2x global_load_b128).
// K-loops are unrolled 2x for software pipelining; epilogue kept rolled to
// hold VGPR usage under 256 for occupancy.
// ---------------------------------------------------------------------------

typedef __attribute__((ext_vector_type(16))) __bf16 v16bf;
typedef __attribute__((ext_vector_type(8)))  __bf16 v8bf;
typedef __attribute__((ext_vector_type(8)))  float  v8f;

constexpr int Nn   = 20000;
constexpr int Ee   = 240000;
constexpr int Bb   = 16;
constexpr int Dd   = 240;
constexpr int Hh   = 4;
constexpr int Ss   = 9;
constexpr int NBr  = 128;
constexpr int MSGd = 2 * Dd + NBr;   // 608 = 19 * 32
constexpr int FEATd = 512;
constexpr int LINI = Hh * Dd + Dd;   // 1200
constexpr int KPAD = 1216;           // 38 * 32
constexpr int KC_MSG = MSGd / 32;    // 19
constexpr int KC_LIN = KPAD / 32;    // 38

static __device__ __forceinline__ float siluf(float x) { return x / (1.0f + expf(-x)); }

// Monotone encoding of float into uint for atomicMax-based segment max.
static __device__ __forceinline__ unsigned encf(float x) {
    unsigned u = __float_as_uint(x);
    return (u & 0x80000000u) ? ~u : (u | 0x80000000u);
}
static __device__ __forceinline__ float decf(unsigned u) {
    unsigned f = (u & 0x80000000u) ? (u & 0x7fffffffu) : ~u;
    return __uint_as_float(f);
}

// ---------------------------------------------------------------------------
// Utility fills
// ---------------------------------------------------------------------------
__global__ void fill_f32_kernel(float* p, float v, int n) {
    int i = blockIdx.x * 256 + threadIdx.x;
    if (i < n) p[i] = v;
}
__global__ void fill_u32_kernel(unsigned* p, unsigned v, int n) {
    int i = blockIdx.x * 256 + threadIdx.x;
    if (i < n) p[i] = v;
}

// ---------------------------------------------------------------------------
// Repack a f32 weight matrix W (K x Nout, row-major) into bf16 WMMA B-tile
// layout: dst[((nt*kcCount + kc)*32 + lane)*16 + j], where
//   col = nt*16 + (lane&15), k = kc*32 + (lane>>4)*8 + (j<8 ? j : j+8), zero-padded k>=K.
// Each thread emits one lane-fragment (16 bf16 = 32 bytes, contiguous).
// ---------------------------------------------------------------------------
__global__ void repack_w_kernel(const float* __restrict__ src, __bf16* __restrict__ dst,
                                int K, int Nout, int kcCount) {
    int ntiles = Nout >> 4;
    int total = ntiles * kcCount * 32;
    int id = blockIdx.x * 256 + threadIdx.x;
    if (id >= total) return;
    int lane = id & 31;
    int t = id >> 5;
    int kc = t % kcCount;
    int nt = t / kcCount;
    int col = nt * 16 + (lane & 15);
    int kbase = kc * 32 + (lane >> 4) * 8;
    __bf16* out = &dst[(size_t)id * 16];
#pragma unroll
    for (int j = 0; j < 16; ++j) {
        int k = kbase + (j < 8 ? j : j + 8);
        out[j] = (k < K) ? (__bf16)src[(size_t)k * Nout + col] : (__bf16)0.f;
    }
}

// ---------------------------------------------------------------------------
// K1: node = embed_tbl[node_atom] @ w_emb2node   (N x 240)
// ---------------------------------------------------------------------------
__global__ void embed_kernel(const int* __restrict__ atom,
                             const float* __restrict__ tbl,     // 64 x 32
                             const float* __restrict__ wproj,   // 32 x 240
                             float* __restrict__ nodef) {
    int i = blockIdx.x * 256 + threadIdx.x;
    if (i >= Nn * Dd) return;
    int n = i / Dd, d = i - n * Dd;
    const float* row = &tbl[atom[n] * 32];
    float s = 0.f;
#pragma unroll
    for (int k = 0; k < 32; ++k) s += row[k] * wproj[k * Dd + d];
    nodef[i] = s;
}

// ---------------------------------------------------------------------------
// K2: e0 MLP on node[:, :64]: 64->64->64->1. One block (64 thr) per node.
// ---------------------------------------------------------------------------
__global__ __launch_bounds__(64) void e0_kernel(const float* __restrict__ nodef,
                                                const float* __restrict__ e1w, const float* __restrict__ e1b,
                                                const float* __restrict__ e2w, const float* __restrict__ e2b,
                                                const float* __restrict__ e3w, const float* __restrict__ e3b,
                                                float* __restrict__ e0out) {
    __shared__ float t[64];
    __shared__ float t2[64];
    int n = blockIdx.x, j = threadIdx.x;
    const float* x = &nodef[(size_t)n * Dd];
    float s = e1b[j];
    for (int k = 0; k < 64; ++k) s += x[k] * e1w[k * 64 + j];
    t[j] = siluf(s);
    __syncthreads();
    float s2 = e2b[j];
    for (int k = 0; k < 64; ++k) s2 += t[k] * e2w[k * 64 + j];
    t2[j] = siluf(s2);
    __syncthreads();
    if (j == 0) {
        float o = e3b[0];
        for (int k = 0; k < 64; ++k) o += t2[k] * e3w[k];
        e0out[n] = o;
    }
}

// ---------------------------------------------------------------------------
// K3: per-edge geometry: rbf (E x 128) and spherical harmonics sh (E x 9)
// ---------------------------------------------------------------------------
__global__ void edge_geom_kernel(const float* __restrict__ evec,
                                 float* __restrict__ rbf,
                                 float* __restrict__ sh) {
    int e = blockIdx.x * 256 + threadIdx.x;
    if (e >= Ee) return;
    float v0 = evec[e * 3 + 0], v1 = evec[e * 3 + 1], v2 = evec[e * 3 + 2];
    float len = sqrtf(v0 * v0 + v1 * v1 + v2 * v2);
    float inv = 1.0f / (len + 1e-12f);
    float x = v1 * inv, y = v2 * inv, z = v0 * inv;   // xr = vec[[1,2,0]]
    const float C3 = 1.7320508075688772f;
    const float C5 = 2.23606797749979f;
    const float C15 = 3.872983346207417f;
    float* shp = &sh[(size_t)e * Ss];
    shp[0] = 1.0f;
    shp[1] = C3 * x;
    shp[2] = C3 * y;
    shp[3] = C3 * z;
    shp[4] = C15 * x * z;
    shp[5] = C15 * x * y;
    shp[6] = 0.5f * C5 * (3.0f * y * y - 1.0f);
    shp[7] = C15 * y * z;
    shp[8] = 0.5f * C15 * (z * z - x * x);
    const float width = 0.5f * 10.0f / (float)NBr;           // 5/128
    const float inv2w2 = 1.0f / (2.0f * width * width);
    float* rp = &rbf[(size_t)e * NBr];
#pragma unroll 4
    for (int i = 0; i < NBr; ++i) {
        float c = 10.0f * (float)i / 127.0f;
        float d = len - c;
        rp[i] = expf(-d * d * inv2w2);
    }
}

// ---------------------------------------------------------------------------
// K4 (per layer): rbf filter MLP (->wv->chs) and attention MLP (->logits).
// One block of 64 threads per edge; LayerNorm reduced through LDS.
// ---------------------------------------------------------------------------
__global__ __launch_bounds__(64) void edge_filter_kernel(
    const float* __restrict__ rbf, const float* __restrict__ sh,
    const float* __restrict__ fc1w, const float* __restrict__ fc1b,
    const float* __restrict__ fc2w, const float* __restrict__ fc2b,
    const float* __restrict__ fc3w, const float* __restrict__ fc3b,
    const float* __restrict__ a1w, const float* __restrict__ a1b,
    const float* __restrict__ a2w, const float* __restrict__ a2b,
    const float* __restrict__ a3w, const float* __restrict__ a3b,
    float* __restrict__ chs, float* __restrict__ logits) {
    __shared__ float r[128];
    __shared__ float t[64];
    __shared__ float t2[64];
    __shared__ float red[2];
    int e = blockIdx.x, j = threadIdx.x;
    r[j]      = rbf[(size_t)e * NBr + j];
    r[j + 64] = rbf[(size_t)e * NBr + 64 + j];
    __syncthreads();

    // fc path: 128 -> 64 -> 64 -> 36 (wv), chs = sh[s] * wv[s][h]
    float s1 = fc1b[j];
    for (int k = 0; k < 128; ++k) s1 += r[k] * fc1w[k * 64 + j];
    t[j] = siluf(s1);
    __syncthreads();
    float s2 = fc2b[j];
    for (int k = 0; k < 64; ++k) s2 += t[k] * fc2w[k * 64 + j];
    t2[j] = siluf(s2);
    __syncthreads();
    if (j < Ss * Hh) {
        float o = fc3b[j];
        for (int k = 0; k < 64; ++k) o += t2[k] * fc3w[k * (Ss * Hh) + j];
        int sidx = j >> 2;  // j = s*H + h
        chs[(size_t)e * (Ss * Hh) + j] = sh[(size_t)e * Ss + sidx] * o;
    }
    __syncthreads();

    // attention path: silu(LN(rbf@a1+b)) -> silu(LN(@a2+b)) -> @a3+b
    float a1 = a1b[j];
    for (int k = 0; k < 128; ++k) a1 += r[k] * a1w[k * 64 + j];
    t[j] = a1;
    __syncthreads();
    if (j == 0) {
        float m = 0.f;
        for (int k = 0; k < 64; ++k) m += t[k];
        m *= (1.0f / 64.0f);
        float v = 0.f;
        for (int k = 0; k < 64; ++k) { float d = t[k] - m; v += d * d; }
        v *= (1.0f / 64.0f);
        red[0] = m; red[1] = rsqrtf(v + 1e-6f);
    }
    __syncthreads();
    t2[j] = siluf((a1 - red[0]) * red[1]);
    __syncthreads();
    float a2 = a2b[j];
    for (int k = 0; k < 64; ++k) a2 += t2[k] * a2w[k * 64 + j];
    t[j] = a2;
    __syncthreads();
    if (j == 0) {
        float m = 0.f;
        for (int k = 0; k < 64; ++k) m += t[k];
        m *= (1.0f / 64.0f);
        float v = 0.f;
        for (int k = 0; k < 64; ++k) { float d = t[k] - m; v += d * d; }
        v *= (1.0f / 64.0f);
        red[0] = m; red[1] = rsqrtf(v + 1e-6f);
    }
    __syncthreads();
    t2[j] = siluf((a2 - red[0]) * red[1]);
    __syncthreads();
    if (j < Hh) {
        float lg = a3b[j];
        for (int k = 0; k < 64; ++k) lg += t2[k] * a3w[k * Hh + j];
        logits[(size_t)e * Hh + j] = lg;
    }
}

// ---------------------------------------------------------------------------
// Segment softmax over edge_dst (3 passes with atomics)
// ---------------------------------------------------------------------------
__global__ void segmax_kernel(const float* __restrict__ logits, const int* __restrict__ edst,
                              unsigned* __restrict__ gmax) {
    int i = blockIdx.x * 256 + threadIdx.x;
    if (i >= Ee * Hh) return;
    int e = i >> 2, h = i & 3;
    atomicMax(&gmax[(size_t)edst[e] * Hh + h], encf(logits[i]));
}
__global__ void exden_kernel(const float* __restrict__ logits, const int* __restrict__ edst,
                             const unsigned* __restrict__ gmax,
                             float* __restrict__ exb, float* __restrict__ den) {
    int i = blockIdx.x * 256 + threadIdx.x;
    if (i >= Ee * Hh) return;
    int e = i >> 2, h = i & 3;
    float g = decf(gmax[(size_t)edst[e] * Hh + h]);
    float ex = expf(logits[i] - g);
    exb[i] = ex;
    atomicAdd(&den[(size_t)edst[e] * Hh + h], ex);
}
__global__ void alpha_kernel(const int* __restrict__ edst, const float* __restrict__ den,
                             float* __restrict__ alpha) {
    int i = blockIdx.x * 256 + threadIdx.x;
    if (i >= Ee * Hh) return;
    int e = i >> 2, h = i & 3;
    alpha[i] = alpha[i] / (den[(size_t)edst[e] * Hh + h] + 1e-12f);
}

// ---------------------------------------------------------------------------
// Build bf16 A fragment from an LDS row: elements [kb..kb+7] ++ [kb+16..kb+23]
// as two 16B vector reads (ds_load_b128).
// ---------------------------------------------------------------------------
static __device__ __forceinline__ v16bf load_afrag(const __bf16* rp) {
    v8bf lo = *(const v8bf*)rp;
    v8bf hi = *(const v8bf*)(rp + 16);
    return __builtin_shufflevector(lo, hi, 0, 1, 2, 3, 4, 5, 6, 7,
                                   8, 9, 10, 11, 12, 13, 14, 15);
}

// ---------------------------------------------------------------------------
// K10: edge message GEMM (E x 608) @ (608 x 240) with WMMA bf16, fused
// spherical/attention epilogue + atomic segment-sum scatter to node_fea.
// Block = 256 threads (8 waves), one block per 16-edge tile.
// B is pre-tiled: each lane fragment is one contiguous 32B load.
// ---------------------------------------------------------------------------
__global__ __launch_bounds__(256) void msg_kernel(
    const __bf16* __restrict__ prew_t,   // tiled: [nt][kc][lane][16]
    const float* __restrict__ nodef,     // N x 240
    const float* __restrict__ rbf,       // E x 128
    const float* __restrict__ chs,       // E x 36
    const float* __restrict__ alpha,     // E x 4
    const float* __restrict__ wtp_l,     // 9 x 240
    const int* __restrict__ esrc, const int* __restrict__ edst,
    float* __restrict__ node_fea)        // N x 960 (atomic accumulate)
{
    __shared__ __bf16 xs[16][MSGd];
    const int e0 = blockIdx.x * 16;
    const int tid = threadIdx.x;

    // Stage concat([node[src], node[dst], rbf]) tile, f32 -> bf16.
    for (int idx = tid; idx < 16 * MSGd; idx += 256) {
        int m = idx / MSGd, k = idx - m * MSGd;
        int e = e0 + m;
        float v;
        if (k < Dd)            v = nodef[(size_t)esrc[e] * Dd + k];
        else if (k < 2 * Dd)   v = nodef[(size_t)edst[e] * Dd + (k - Dd)];
        else                   v = rbf[(size_t)e * NBr + (k - 2 * Dd)];
        xs[m][k] = (__bf16)v;
    }
    __syncthreads();

    const int wave = tid >> 5;
    const int lane = tid & 31;
    const int row16 = lane & 15;
    const int hi = lane >> 4;

    for (int nt = wave; nt < Dd / 16; nt += 8) {
        const int col = nt * 16 + row16;
        const __bf16* bt = &prew_t[((size_t)nt * KC_MSG * 32 + lane) * 16];
        v8f acc = {0.f, 0.f, 0.f, 0.f, 0.f, 0.f, 0.f, 0.f};
        // 2x unrolled K loop (19 chunks): pipeline loads across WMMAs.
#pragma unroll 2
        for (int kc = 0; kc < KC_MSG; ++kc) {
            v16bf a = load_afrag(&xs[row16][kc * 32 + hi * 8]);
            v16bf b = *(const v16bf*)(bt + (size_t)kc * 32 * 16);
            __builtin_prefetch(bt + (size_t)(kc + 1) * 32 * 16, 0, 1);
            acc = __builtin_amdgcn_wmma_f32_16x16x32_bf16(false, a, false, b,
                                                          (short)0, acc, false, false);
        }
        // Epilogue: edge_out[e,h,col] = msg * alpha[e,h] * sum_s chs[e,s,h]*wtp[s,col]
        float wt[Ss];
#pragma unroll
        for (int s = 0; s < Ss; ++s) wt[s] = wtp_l[s * Dd + col];
        // Keep rolled: bounds VGPR usage (occupancy) at the cost of a tiny loop.
#pragma unroll 1
        for (int j = 0; j < 8; ++j) {
            int e = e0 + hi * 8 + j;
            int dnode = edst[e];
            float mv = acc[j];
            const float4* cp = (const float4*)&chs[(size_t)e * (Ss * Hh)];
            float g0 = 0.f, g1 = 0.f, g2 = 0.f, g3 = 0.f;
#pragma unroll
            for (int s = 0; s < Ss; ++s) {
                float4 c4 = cp[s];
                g0 += c4.x * wt[s];
                g1 += c4.y * wt[s];
                g2 += c4.z * wt[s];
                g3 += c4.w * wt[s];
            }
            float4 al = *(const float4*)&alpha[(size_t)e * Hh];
            float* np = &node_fea[(size_t)dnode * (Hh * Dd) + col];
            atomicAdd(np + 0 * Dd, mv * al.x * g0);
            atomicAdd(np + 1 * Dd, mv * al.y * g1);
            atomicAdd(np + 2 * Dd, mv * al.z * g2);
            atomicAdd(np + 3 * Dd, mv * al.w * g3);
        }
    }
}

// ---------------------------------------------------------------------------
// K11: node update GEMM: concat([node_fea(960), node(240)]) @ lw (1200 x Dout)
// WMMA bf16, K padded 1200->1216 (padding baked into tiled B).
// Block = 256 threads per 16-node tile.
// ---------------------------------------------------------------------------
__global__ __launch_bounds__(256) void node_update_kernel(
    const __bf16* __restrict__ lw_t,     // tiled: [nt][kc][lane][16]
    const float* __restrict__ node_fea,  // N x 960
    const float* __restrict__ node_in,   // N x 240
    float* __restrict__ node_out,        // N x Dout
    int Dout)
{
    __shared__ __bf16 xs[16][KPAD];
    const int g0 = blockIdx.x * 16;
    const int tid = threadIdx.x;
    for (int idx = tid; idx < 16 * KPAD; idx += 256) {
        int m = idx / KPAD, k = idx - m * KPAD;
        int g = g0 + m;
        float v = 0.f;
        if (k < Hh * Dd)       v = node_fea[(size_t)g * (Hh * Dd) + k];
        else if (k < LINI)     v = node_in[(size_t)g * Dd + (k - Hh * Dd)];
        xs[m][k] = (__bf16)v;
    }
    __syncthreads();

    const int wave = tid >> 5;
    const int lane = tid & 31;
    const int row16 = lane & 15;
    const int hi = lane >> 4;
    const int ntiles = Dout >> 4;

    for (int nt = wave; nt < ntiles; nt += 8) {
        const int col = nt * 16 + row16;
        const __bf16* bt = &lw_t[((size_t)nt * KC_LIN * 32 + lane) * 16];
        v8f acc = {0.f, 0.f, 0.f, 0.f, 0.f, 0.f, 0.f, 0.f};
#pragma unroll 2
        for (int kc = 0; kc < KC_LIN; ++kc) {
            v16bf a = load_afrag(&xs[row16][kc * 32 + hi * 8]);
            v16bf b = *(const v16bf*)(bt + (size_t)kc * 32 * 16);
            __builtin_prefetch(bt + (size_t)(kc + 1) * 32 * 16, 0, 1);
            acc = __builtin_amdgcn_wmma_f32_16x16x32_bf16(false, a, false, b,
                                                          (short)0, acc, false, false);
        }
#pragma unroll
        for (int j = 0; j < 8; ++j)
            node_out[(size_t)(g0 + hi * 8 + j) * Dout + col] = acc[j];
    }
}

// ---------------------------------------------------------------------------
// K12: head MLP 512->64->64->1 (+e0), atomic segment-sum into energy[batch]
// ---------------------------------------------------------------------------
__global__ __launch_bounds__(64) void head_kernel(const float* __restrict__ nodef,  // N x 512
                                                  const float* __restrict__ h1w, const float* __restrict__ h1b,
                                                  const float* __restrict__ h2w, const float* __restrict__ h2b,
                                                  const float* __restrict__ h3w, const float* __restrict__ h3b,
                                                  const float* __restrict__ e0buf,
                                                  const int* __restrict__ batch,
                                                  float* __restrict__ out) {
    __shared__ float t[64];
    __shared__ float t2[64];
    int n = blockIdx.x, j = threadIdx.x;
    const float* x = &nodef[(size_t)n * FEATd];
    float s = h1b[j];
    for (int k = 0; k < FEATd; ++k) s += x[k] * h1w[k * 64 + j];
    t[j] = siluf(s);
    __syncthreads();
    float s2 = h2b[j];
    for (int k = 0; k < 64; ++k) s2 += t[k] * h2w[k * 64 + j];
    t2[j] = siluf(s2);
    __syncthreads();
    if (j == 0) {
        float o = h3b[0];
        for (int k = 0; k < 64; ++k) o += t2[k] * h3w[k];
        o += e0buf[n];
        atomicAdd(&out[batch[n]], o);
    }
}

// ---------------------------------------------------------------------------
// Host side
// ---------------------------------------------------------------------------
extern "C" void kernel_launch(void* const* d_in, const int* in_sizes, int n_in,
                              void* d_out, int out_size, void* d_ws, size_t ws_size,
                              hipStream_t stream) {
    const int*   node_atom  = (const int*)d_in[0];
    const int*   edge_src   = (const int*)d_in[1];
    const int*   edge_dst   = (const int*)d_in[2];
    const int*   batch      = (const int*)d_in[3];
    const float* edge_vec   = (const float*)d_in[4];
    const float* embed_tbl  = (const float*)d_in[5];
    const float* w_emb2node = (const float*)d_in[6];
    const float* pre_w      = (const float*)d_in[7];
    const float* fc1_w      = (const float*)d_in[8];
    const float* fc1_b      = (const float*)d_in[9];
    const float* fc2_w      = (const float*)d_in[10];
    const float* fc2_b      = (const float*)d_in[11];
    const float* fc3_w      = (const float*)d_in[12];
    const float* fc3_b      = (const float*)d_in[13];
    const float* wtp        = (const float*)d_in[14];
    const float* a1_w       = (const float*)d_in[15];
    const float* a1_b       = (const float*)d_in[16];
    const float* a2_w       = (const float*)d_in[17];
    const float* a2_b       = (const float*)d_in[18];
    const float* a3_w       = (const float*)d_in[19];
    const float* a3_b       = (const float*)d_in[20];
    const float* lin_mid_w  = (const float*)d_in[21];
    const float* lin_last_w = (const float*)d_in[22];
    const float* e1_w       = (const float*)d_in[23];
    const float* e1_b       = (const float*)d_in[24];
    const float* e2_w       = (const float*)d_in[25];
    const float* e2_b       = (const float*)d_in[26];
    const float* e3_w       = (const float*)d_in[27];
    const float* e3_b       = (const float*)d_in[28];
    const float* h1_w       = (const float*)d_in[29];
    const float* h1_b       = (const float*)d_in[30];
    const float* h2_w       = (const float*)d_in[31];
    const float* h2_b       = (const float*)d_in[32];
    const float* h3_w       = (const float*)d_in[33];
    const float* h3_b       = (const float*)d_in[34];

    char* wsp = (char*)d_ws;
    auto alloc = [&](size_t bytes) -> char* {
        char* p = wsp;
        wsp += (bytes + 255) & ~(size_t)255;
        return p;
    };
    // Tiled B sizes (elements): ntiles * kcCount * 32 * 16
    const size_t prewT_sz = (size_t)(Dd / 16) * KC_MSG * 32 * 16;     // per layer
    const size_t linmT_sz = (size_t)(Dd / 16) * KC_LIN * 32 * 16;
    const size_t linlT_sz = (size_t)(FEATd / 16) * KC_LIN * 32 * 16;

    float*    nodeA    = (float*)alloc(sizeof(float) * (size_t)Nn * Dd);
    float*    nodeB    = (float*)alloc(sizeof(float) * (size_t)Nn * Dd);
    float*    nodeF    = (float*)alloc(sizeof(float) * (size_t)Nn * FEATd);
    float*    e0buf    = (float*)alloc(sizeof(float) * (size_t)Nn);
    float*    rbf      = (float*)alloc(sizeof(float) * (size_t)Ee * NBr);
    float*    sh       = (float*)alloc(sizeof(float) * (size_t)Ee * Ss);
    float*    chs      = (float*)alloc(sizeof(float) * (size_t)Ee * Ss * Hh);
    float*    logits   = (float*)alloc(sizeof(float) * (size_t)Ee * Hh);
    float*    alpha    = (float*)alloc(sizeof(float) * (size_t)Ee * Hh);
    unsigned* gmax     = (unsigned*)alloc(sizeof(unsigned) * (size_t)Nn * Hh);
    float*    den      = (float*)alloc(sizeof(float) * (size_t)Nn * Hh);
    float*    node_fea = (float*)alloc(sizeof(float) * (size_t)Nn * Hh * Dd);
    __bf16*   prew_t   = (__bf16*)alloc(sizeof(__bf16) * 2 * prewT_sz);
    __bf16*   linm_t   = (__bf16*)alloc(sizeof(__bf16) * linmT_sz);
    __bf16*   linl_t   = (__bf16*)alloc(sizeof(__bf16) * linlT_sz);

    auto blks = [](long long n) { return (int)((n + 255) / 256); };

    // Repack weights into WMMA lane-tiled bf16 layout.
    repack_w_kernel<<<blks(prewT_sz / 16), 256, 0, stream>>>(pre_w, prew_t, MSGd, Dd, KC_MSG);
    repack_w_kernel<<<blks(prewT_sz / 16), 256, 0, stream>>>(pre_w + (size_t)MSGd * Dd,
                                                             prew_t + prewT_sz, MSGd, Dd, KC_MSG);
    repack_w_kernel<<<blks(linmT_sz / 16), 256, 0, stream>>>(lin_mid_w, linm_t, LINI, Dd, KC_LIN);
    repack_w_kernel<<<blks(linlT_sz / 16), 256, 0, stream>>>(lin_last_w, linl_t, LINI, FEATd, KC_LIN);

    // Node embedding + e0 MLP + edge geometry
    embed_kernel<<<blks((long long)Nn * Dd), 256, 0, stream>>>(node_atom, embed_tbl, w_emb2node, nodeA);
    e0_kernel<<<Nn, 64, 0, stream>>>(nodeA, e1_w, e1_b, e2_w, e2_b, e3_w, e3_b, e0buf);
    edge_geom_kernel<<<blks(Ee), 256, 0, stream>>>(edge_vec, rbf, sh);

    for (int l = 0; l < 2; ++l) {
        const float* nin = (l == 0) ? nodeA : nodeB;
        edge_filter_kernel<<<Ee, 64, 0, stream>>>(
            rbf, sh,
            fc1_w + (size_t)l * NBr * 64, fc1_b + (size_t)l * 64,
            fc2_w + (size_t)l * 64 * 64,  fc2_b + (size_t)l * 64,
            fc3_w + (size_t)l * 64 * Ss * Hh, fc3_b + (size_t)l * Ss * Hh,
            a1_w + (size_t)l * NBr * 64,  a1_b + (size_t)l * 64,
            a2_w + (size_t)l * 64 * 64,   a2_b + (size_t)l * 64,
            a3_w + (size_t)l * 64 * Hh,   a3_b + (size_t)l * Hh,
            chs, logits);

        fill_u32_kernel<<<blks(Nn * Hh), 256, 0, stream>>>(gmax, 0u, Nn * Hh);
        fill_f32_kernel<<<blks(Nn * Hh), 256, 0, stream>>>(den, 0.f, Nn * Hh);
        segmax_kernel<<<blks((long long)Ee * Hh), 256, 0, stream>>>(logits, edge_dst, gmax);
        exden_kernel<<<blks((long long)Ee * Hh), 256, 0, stream>>>(logits, edge_dst, gmax, alpha, den);
        alpha_kernel<<<blks((long long)Ee * Hh), 256, 0, stream>>>(edge_dst, den, alpha);

        fill_f32_kernel<<<blks((long long)Nn * Hh * Dd), 256, 0, stream>>>(node_fea, 0.f, Nn * Hh * Dd);
        msg_kernel<<<Ee / 16, 256, 0, stream>>>(
            prew_t + (size_t)l * prewT_sz, nin, rbf, chs, alpha,
            wtp + (size_t)l * Ss * Dd, edge_src, edge_dst, node_fea);

        if (l == 0) {
            node_update_kernel<<<Nn / 16, 256, 0, stream>>>(linm_t, node_fea, nin, nodeB, Dd);
        } else {
            node_update_kernel<<<Nn / 16, 256, 0, stream>>>(linl_t, node_fea, nin, nodeF, FEATd);
        }
    }

    fill_f32_kernel<<<1, 256, 0, stream>>>((float*)d_out, 0.f, Bb);
    head_kernel<<<Nn, 64, 0, stream>>>(nodeF, h1_w, h1_b, h2_w, h2_b, h3_w, h3_b,
                                       e0buf, batch, (float*)d_out);
}
